// ERENet_55104430408295
// MI455X (gfx1250) — matrix operand order repaired and checked by reading, not compiled, not run on hardware
//
#include <hip/hip_runtime.h>
#include <math.h>

// ---------------------------------------------------------------------------
// ERENet forward (CRF loss + relation-selection BCE) for gfx1250 / MI455X.
// Heavy GEMMs use v_wmma_f32_16x16x32_bf16 (wave32 WMMA) with 2x2 register
// blocking; the sequential LSTM recurrence runs as persistent workgroups.
// ---------------------------------------------------------------------------

typedef __attribute__((ext_vector_type(16))) __bf16        v16bf;
typedef __attribute__((ext_vector_type(8)))  float          v8f;
typedef __attribute__((ext_vector_type(4)))  unsigned int   u32x4;

union FragB16 { v16bf v; u32x4 q[2]; unsigned short u[16]; };

__device__ __forceinline__ unsigned short f2bf(float f) {
  unsigned int u = __builtin_bit_cast(unsigned int, f);
  u += 0x7FFFu + ((u >> 16) & 1u);           // round-to-nearest-even
  return (unsigned short)(u >> 16);
}
__device__ __forceinline__ float sigm(float x) { return 1.f / (1.f + __expf(-x)); }

#define WMMA_BF16(A_, B_, C_) \
  __builtin_amdgcn_wmma_f32_16x16x32_bf16(false, (A_), false, (B_), (short)0, (C_), false, false)

// ---------------- misc init / converts ----------------
__global__ void init_misc_kernel(float* misc) {
  if (threadIdx.x < 2) misc[threadIdx.x] = 0.f;   // [0]=crf_loss, [1]=bce accumulator
}

__global__ void f32_to_bf16_kernel(const float* __restrict__ s,
                                   unsigned short* __restrict__ d, int n) {
  int i = blockIdx.x * 256 + threadIdx.x;
  if (i < n) d[i] = f2bf(s[i]);
}

// src[K][N] f32 -> dst[N][K] bf16
__global__ void transpose_bf16_kernel(const float* __restrict__ s,
                                      unsigned short* __restrict__ d, int K, int N) {
  int i = blockIdx.x * 256 + threadIdx.x;
  if (i < K * N) { int k = i / N, n = i - k * N; d[n * K + k] = f2bf(s[i]); }
}

// rel_table [49][128] f32 -> [64][128] bf16, rows 49..63 zero
__global__ void relprep_kernel(const float* __restrict__ rel, unsigned short* __restrict__ d) {
  int i = blockIdx.x * 256 + threadIdx.x;
  if (i < 64 * 128) { int r = i >> 7; d[i] = (r < 49) ? f2bf(rel[i]) : (unsigned short)0; }
}

// ---------------- embedding: emb[l*16+b] = char_emb + word_emb @ w_wc ----------------
__global__ void embed_kernel(const int* __restrict__ cid, const int* __restrict__ wid,
                             const float* __restrict__ wtab, const float* __restrict__ ctab,
                             const float* __restrict__ wwc, unsigned short* __restrict__ emb) {
  int row = blockIdx.x;                 // l*16 + b (LSTM token order)
  int l = row >> 4, b = row & 15;
  int ci = cid[b * 128 + l], wi = wid[b * 128 + l];
  int ch = threadIdx.x;                 // 128
  float acc = ctab[(size_t)ci * 128 + ch];
  const float* wr = wtab + (size_t)wi * 300;
  for (int w = 0; w < 300; ++w) acc += wr[w] * wwc[w * 128 + ch];
  emb[(size_t)row * 128 + ch] = f2bf(acc);
}

// ---------------- generic bf16 WMMA GEMM, 2x2 register blocking ----------------
// C[M][N] = A[M][K](bf16,row-major) x W[N][K](bf16,row-major, i.e. B^T) (+bias, relu)
// 4 waves/block; each wave owns a 32x32 output block (2x2 WMMA tiles), K chunks of 32.
// Fragment layouts per CDNA5 ISA:
//   A: lane m=lane&15, half=lane>>4; elems 0-7 -> K=k0+half*8+e ; 8-15 -> +16
//   B: lane n=lane&15; elems 0-7 -> K=k0+half*8+e ; 8-15 -> +16 (rows of W = output columns)
//   C: M = vgpr + 8*half, N = lane&15
__launch_bounds__(128)
__global__ void gemm_bf16_kernel(const unsigned short* __restrict__ A,
                                 const unsigned short* __restrict__ W,
                                 const float* __restrict__ bias,
                                 float* __restrict__ Cf, unsigned short* __restrict__ Cb,
                                 int M, int N, int K, int relu) {
  int wave = threadIdx.x >> 5, lane = threadIdx.x & 31;
  int m16 = lane & 15, half = lane >> 4;
  int ntN = N >> 5;                                   // 32-wide macro column blocks
  int mt = blockIdx.x * 4 + wave;
  if (mt >= (M >> 5) * ntN) return;
  int tm = mt / ntN, tn = mt - tm * ntN;
  const unsigned short* Arow0 = A + (size_t)(tm * 32 + m16) * K;
  const unsigned short* Arow1 = Arow0 + (size_t)16 * K;
  const unsigned short* Wrow0 = W + (size_t)(tn * 32 + m16) * K;
  const unsigned short* Wrow1 = Wrow0 + (size_t)16 * K;
  v8f acc00 = {}, acc01 = {}, acc10 = {}, acc11 = {};
  for (int k0 = 0; k0 < K; k0 += 32) {
    FragB16 a0, a1, b0, b1;
    a0.q[0] = *(const u32x4*)(Arow0 + k0 + half * 8);
    a0.q[1] = *(const u32x4*)(Arow0 + k0 + 16 + half * 8);
    a1.q[0] = *(const u32x4*)(Arow1 + k0 + half * 8);
    a1.q[1] = *(const u32x4*)(Arow1 + k0 + 16 + half * 8);
    b0.q[0] = *(const u32x4*)(Wrow0 + k0 + half * 8);
    b0.q[1] = *(const u32x4*)(Wrow0 + k0 + 16 + half * 8);
    b1.q[0] = *(const u32x4*)(Wrow1 + k0 + half * 8);
    b1.q[1] = *(const u32x4*)(Wrow1 + k0 + 16 + half * 8);
    __builtin_prefetch(Arow0 + k0 + 128, 0, 1);
    __builtin_prefetch(Arow1 + k0 + 128, 0, 1);
    acc00 = WMMA_BF16(a0.v, b0.v, acc00);
    acc01 = WMMA_BF16(a0.v, b1.v, acc01);
    acc10 = WMMA_BF16(a1.v, b0.v, acc10);
    acc11 = WMMA_BF16(a1.v, b1.v, acc11);
  }
  int col0 = tn * 32 + m16, col1 = col0 + 16;
  float bv0 = bias ? bias[col0] : 0.f;
  float bv1 = bias ? bias[col1] : 0.f;
#pragma unroll
  for (int v = 0; v < 8; ++v) {
    int row0 = tm * 32 + v + half * 8;
    int row1 = row0 + 16;
    float x00 = acc00[v] + bv0, x01 = acc01[v] + bv1;
    float x10 = acc10[v] + bv0, x11 = acc11[v] + bv1;
    if (relu) {
      x00 = fmaxf(x00, 0.f); x01 = fmaxf(x01, 0.f);
      x10 = fmaxf(x10, 0.f); x11 = fmaxf(x11, 0.f);
    }
    if (Cf) {
      Cf[(size_t)row0 * N + col0] = x00; Cf[(size_t)row0 * N + col1] = x01;
      Cf[(size_t)row1 * N + col0] = x10; Cf[(size_t)row1 * N + col1] = x11;
    }
    if (Cb) {
      Cb[(size_t)row0 * N + col0] = f2bf(x00); Cb[(size_t)row0 * N + col1] = f2bf(x01);
      Cb[(size_t)row1 * N + col0] = f2bf(x10); Cb[(size_t)row1 * N + col1] = f2bf(x11);
    }
  }
}

// ---------------- bidirectional LSTM layer (persistent, 1 WG per direction) ----------
// gates g[b,1024] = xW[pos,b,:] + bias + h @ Whh^T ; pointwise LSTM update.
__launch_bounds__(1024)
__global__ void lstm_layer_kernel(const float* __restrict__ xW0, const float* __restrict__ xW1,
                                  const float* __restrict__ Whh0, const float* __restrict__ Whh1,
                                  const float* __restrict__ b0, const float* __restrict__ b1,
                                  float* __restrict__ xnext) {
  __shared__ float h[4096], c[4096], g4[4096];   // 16x256, 16x256, 4x1024 = 48KB
  int dir = blockIdx.x;
  const float* xW   = dir ? xW1  : xW0;
  const float* Whh  = dir ? Whh1 : Whh0;
  const float* bias = dir ? b1   : b0;
  int tid = threadIdx.x;
  for (int idx = tid; idx < 4096; idx += 1024) { h[idx] = 0.f; c[idx] = 0.f; }
  __syncthreads();
  float bj = bias[tid];
  const float* wrow = Whh + (size_t)tid * 256;
  for (int t = 0; t < 128; ++t) {
    int pos = dir ? (127 - t) : t;
    float acc[16];
#pragma unroll
    for (int b = 0; b < 16; ++b) acc[b] = xW[(size_t)(pos * 16 + b) * 1024 + tid] + bj;
#pragma unroll 4
    for (int k = 0; k < 256; ++k) {
      float wv = wrow[k];
#pragma unroll
      for (int b = 0; b < 16; ++b) acc[b] += h[b * 256 + k] * wv;  // LDS broadcast
    }
#pragma unroll
    for (int q = 0; q < 4; ++q) {       // stage 4 batches at a time through LDS
#pragma unroll
      for (int b = 0; b < 4; ++b) g4[b * 1024 + tid] = acc[q * 4 + b];
      __syncthreads();
      if (tid < 256) {
        int k = tid;
#pragma unroll
        for (int bb = 0; bb < 4; ++bb) {
          int b = q * 4 + bb;
          float ii = sigm(g4[bb * 1024 + k]);
          float ff = sigm(g4[bb * 1024 + 256 + k]);
          float gg = tanhf(g4[bb * 1024 + 512 + k]);
          float oo = sigm(g4[bb * 1024 + 768 + k]);
          float cc = ff * c[b * 256 + k] + ii * gg;
          float hh = oo * tanhf(cc);
          c[b * 256 + k] = cc; h[b * 256 + k] = hh;
          xnext[(size_t)(pos * 16 + b) * 512 + dir * 256 + k] = hh;
        }
      }
      __syncthreads();
    }
  }
}

// ---------------- emission: e[u][j] = sent[u] . em_W[:,j] + em_b[j] ----------------
__global__ void emission_kernel(const float* __restrict__ x2, const float* __restrict__ emW,
                                const float* __restrict__ emBias, float* __restrict__ e) {
  int u = blockIdx.x * 16 + (threadIdx.x >> 4);   // u = b*128 + l
  int j = threadIdx.x & 15;
  int b = u >> 7, l = u & 127;
  const float* s = x2 + (size_t)(l * 16 + b) * 512;
  float acc = 0.f;
  if (j < 15) {
    acc = emBias[j];
    for (int k = 0; k < 512; ++k) acc += s[k] * emW[k * 15 + j];
  }
  e[u * 16 + j] = acc;
}

// ---------------- CRF: numerator + forward algorithm -> misc[0] ----------------
__launch_bounds__(256)
__global__ void crf_kernel(const float* __restrict__ e, const int* __restrict__ tags,
                           const float* __restrict__ cs, const float* __restrict__ ce,
                           const float* __restrict__ ct, float* __restrict__ misc) {
  __shared__ float tr[225], sst[15], sen[15];
  __shared__ float sc[240], nw[240], snum[16], slogZ[16];
  int tid = threadIdx.x;
  if (tid < 225) tr[tid] = ct[tid];
  if (tid < 15) { sst[tid] = cs[tid]; sen[tid] = ce[tid]; }
  __syncthreads();
  if (tid < 16) {
    int b = tid; const int* tg = tags + b * 128;
    float a = sst[tg[0]] + sen[tg[127]];
    for (int l = 0; l < 128; ++l) a += e[(b * 128 + l) * 16 + tg[l]];
    for (int l = 1; l < 128; ++l) a += tr[tg[l - 1] * 15 + tg[l]];
    snum[b] = a;
  }
  if (tid < 240) { int b = tid / 15, j = tid - b * 15; sc[tid] = sst[j] + e[(b * 128) * 16 + j]; }
  __syncthreads();
  for (int t = 1; t < 128; ++t) {
    if (tid < 240) {
      int b = tid / 15, j = tid - b * 15;
      float m = -1e30f;
      for (int k = 0; k < 15; ++k) m = fmaxf(m, sc[b * 15 + k] + tr[k * 15 + j]);
      float s = 0.f;
      for (int k = 0; k < 15; ++k) s += __expf(sc[b * 15 + k] + tr[k * 15 + j] - m);
      nw[tid] = m + __logf(s) + e[(b * 128 + t) * 16 + j];
    }
    __syncthreads();
    if (tid < 240) sc[tid] = nw[tid];
    __syncthreads();
  }
  if (tid < 16) {
    int b = tid; float m = -1e30f;
    for (int j = 0; j < 15; ++j) m = fmaxf(m, sc[b * 15 + j] + sen[j]);
    float s = 0.f;
    for (int j = 0; j < 15; ++j) s += __expf(sc[b * 15 + j] + sen[j] - m);
    slogZ[b] = m + __logf(s);
  }
  __syncthreads();
  if (tid == 0) {
    float a = 0.f;
    for (int b = 0; b < 16; ++b) a += snum[b] - slogZ[b];
    misc[0] = -a / 16.f;
  }
}

// ---------------- rel_enc = concat(sent, ent_table[labels]) in bf16 ----------------
__global__ void relenc_kernel(const float* __restrict__ x2, const float* __restrict__ ent,
                              const int* __restrict__ tags, unsigned short* __restrict__ renc) {
  int i = blockIdx.x * 256 + threadIdx.x;
  if (i >= 2048 * 576) return;
  int u = i / 576, f = i - u * 576;
  int b = u >> 7, l = u & 127;
  float val = (f < 512) ? x2[(size_t)(l * 16 + b) * 512 + f]
                        : ent[tags[b * 128 + l] * 64 + (f - 512)];
  renc[i] = f2bf(val);
}

// ---------------- fused UV + logits (WMMA) + BCE, one block per (b,i) ----------------
__launch_bounds__(256)
__global__ void rel_fused_kernel(const float* __restrict__ uWa, const float* __restrict__ vWb,
                                 const float* __restrict__ sb, const unsigned short* __restrict__ relW,
                                 const float* __restrict__ spo, float* __restrict__ sel_acc) {
  __shared__ unsigned short sUV[128 * 128];   // 32KB
  __shared__ unsigned short sRel[64 * 128];   // 16KB
  __shared__ float sRed[256];
  int tid = threadIdx.x;
  int u = blockIdx.x;                         // u = b*128 + i
  int b = u >> 7;
  const float* vrow = vWb + (size_t)u * 128;
  // UV[j][h] = relu(uWa[b,j,h] + vWb[b,i,h] + suv_b[h])
  for (int idx = tid; idx < 128 * 128; idx += 256) {
    int j = idx >> 7, h = idx & 127;
    float val = uWa[(size_t)(b * 128 + j) * 128 + h] + vrow[h] + sb[h];
    sUV[idx] = f2bf(fmaxf(val, 0.f));
  }
  for (int idx = tid; idx < 64 * 128; idx += 256) sRel[idx] = relW[idx];
  __syncthreads();
  // logits[j][r] = sum_h UV[j][h] * rel[r][h] : 8 waves x (4 r-tiles) x (4 K-chunks)
  int wave = tid >> 5, lane = tid & 31;
  int m16 = lane & 15, half = lane >> 4;
  int tm = wave;
  v8f acc0 = {}, acc1 = {}, acc2 = {}, acc3 = {};
  const unsigned short* Arow = sUV + (tm * 16 + m16) * 128;
  const unsigned short* W0 = sRel + (0 * 16 + m16) * 128;
  const unsigned short* W1 = sRel + (1 * 16 + m16) * 128;
  const unsigned short* W2 = sRel + (2 * 16 + m16) * 128;
  const unsigned short* W3 = sRel + (3 * 16 + m16) * 128;
#pragma unroll
  for (int k0 = 0; k0 < 128; k0 += 32) {
    FragB16 a, w;
    a.q[0] = *(const u32x4*)(Arow + k0 + half * 8);
    a.q[1] = *(const u32x4*)(Arow + k0 + 16 + half * 8);
    w.q[0] = *(const u32x4*)(W0 + k0 + half * 8);
    w.q[1] = *(const u32x4*)(W0 + k0 + 16 + half * 8);
    acc0 = WMMA_BF16(a.v, w.v, acc0);
    w.q[0] = *(const u32x4*)(W1 + k0 + half * 8);
    w.q[1] = *(const u32x4*)(W1 + k0 + 16 + half * 8);
    acc1 = WMMA_BF16(a.v, w.v, acc1);
    w.q[0] = *(const u32x4*)(W2 + k0 + half * 8);
    w.q[1] = *(const u32x4*)(W2 + k0 + 16 + half * 8);
    acc2 = WMMA_BF16(a.v, w.v, acc2);
    w.q[0] = *(const u32x4*)(W3 + k0 + half * 8);
    w.q[1] = *(const u32x4*)(W3 + k0 + 16 + half * 8);
    acc3 = WMMA_BF16(a.v, w.v, acc3);
  }
  float lsum = 0.f;
  const float* spoB = spo + (size_t)u * 49 * 128;    // spo[b,i,:,:]
#pragma unroll
  for (int tn = 0; tn < 4; ++tn) {
    v8f acc = (tn == 0) ? acc0 : (tn == 1) ? acc1 : (tn == 2) ? acc2 : acc3;
    int r = tn * 16 + m16;
    if (r < 49) {
#pragma unroll
      for (int v = 0; v < 8; ++v) {
        int j = tm * 16 + v + half * 8;
        float x = acc[v];
        float z = spoB[r * 128 + j];
        lsum += fmaxf(x, 0.f) - x * z + log1pf(__expf(-fabsf(x)));
      }
    }
  }
  sRed[tid] = lsum; __syncthreads();
  for (int s = 128; s > 0; s >>= 1) { if (tid < s) sRed[tid] += sRed[tid + s]; __syncthreads(); }
  if (tid == 0) atomicAdd(sel_acc, sRed[0]);
}

__global__ void combine_kernel(const float* __restrict__ misc, float* __restrict__ out) {
  out[0] = misc[0] + misc[1] * (1.f / 2048.f);
}

// ---------------------------------------------------------------------------
extern "C" void kernel_launch(void* const* d_in, const int* in_sizes, int n_in,
                              void* d_out, int out_size, void* d_ws, size_t ws_size,
                              hipStream_t stream) {
  (void)n_in; (void)out_size; (void)ws_size;
  const int*   char_ids = (const int*)d_in[0];
  const int*   word_ids = (const int*)d_in[1];
  const int*   label_ids= (const int*)d_in[2];
  const float* spo      = (const float*)d_in[3];
  const float* wtab     = (const float*)d_in[4];
  const float* ctab     = (const float*)d_in[5];
  const float* enttab   = (const float*)d_in[6];
  const float* reltab   = (const float*)d_in[7];
  const float* wwc      = (const float*)d_in[8];

  // lstm params at indices 9..20; resolve Wih/Whh/b per (layer,dir) by size.
  const float *Wih[2][2] = {}, *Whh[2][2] = {}, *Bg[2][2] = {};
  int idx = 9;
  for (int l = 0; l < 2; ++l)
    for (int d = 0; d < 2; ++d) {
      long din = (l == 0) ? 128 : 512;
      for (int k = 0; k < 3; ++k) {
        const float* p = (const float*)d_in[idx + k];
        long sz = in_sizes[idx + k];
        if (sz == 1024) Bg[l][d] = p;
        else if (sz == 1024 * din) Wih[l][d] = p;
        else Whh[l][d] = p;
      }
      idx += 3;
    }
  const float* emW  = (const float*)d_in[21];
  const float* emb_b= (const float*)d_in[22];
  const float* suW  = (const float*)d_in[23];
  const float* sub  = (const float*)d_in[24];
  const float* svW  = (const float*)d_in[25];
  const float* svb  = (const float*)d_in[26];
  const float* suvW = (const float*)d_in[27];
  const float* suvb = (const float*)d_in[28];
  const float* crs  = (const float*)d_in[29];
  const float* cre  = (const float*)d_in[30];
  const float* crt  = (const float*)d_in[31];

  char* w = (char*)d_ws;
  unsigned short* EMB  = (unsigned short*)(w + 0);              // 2048x128 bf16
  float*          XW0  = (float*)(w + 524288);                  // 2048x1024 f32
  float*          XW1  = (float*)(w + 8912896);                 // 2048x1024 f32
  float*          X1   = (float*)(w + 17301504);                // 2048x512 f32
  unsigned short* X1B  = (unsigned short*)(w + 21495808);       // 2048x512 bf16
  float*          X2   = (float*)(w + 23592960);                // 2048x512 f32
  unsigned short* WCA  = (unsigned short*)(w + 27787264);       // <=1024x512 bf16
  unsigned short* WCB  = (unsigned short*)(w + 28835840);
  float*          EMIS = (float*)(w + 29884416);                // 2048x16 f32
  float*          MISC = (float*)(w + 30015488);                // accumulators
  // rel-stage buffers aliased into the (then free) XW region:
  unsigned short* RENC = (unsigned short*)(w + 524288);         // 2048x576 bf16
  unsigned short* SUWT = (unsigned short*)(w + 2883584);        // 128x576
  unsigned short* SVWT = (unsigned short*)(w + 3031040);        // 128x576
  unsigned short* WAT  = (unsigned short*)(w + 3178496);        // 128x128
  unsigned short* WBT  = (unsigned short*)(w + 3211264);        // 128x128
  unsigned short* RELW = (unsigned short*)(w + 3244032);        // 64x128
  unsigned short* UB   = (unsigned short*)(w + 3260416);        // 2048x128
  unsigned short* VB   = (unsigned short*)(w + 3784704);        // 2048x128
  float*          UWA  = (float*)(w + 4308992);                 // 2048x128 f32
  float*          VWB  = (float*)(w + 5357568);                 // 2048x128 f32

  init_misc_kernel<<<1, 32, 0, stream>>>(MISC);
  embed_kernel<<<2048, 128, 0, stream>>>(char_ids, word_ids, wtab, ctab, wwc, EMB);

  // grid for 32x32-blocked GEMM: ((M/32)*(N/32)) / 4 waves per block
  // M=2048,N=1024 -> 2048 macro tiles -> 512 blocks ; M=2048,N=128 -> 256 -> 64 blocks

  // ---- LSTM layer 0 ----
  f32_to_bf16_kernel<<<512, 256, 0, stream>>>(Wih[0][0], WCA, 1024 * 128);
  f32_to_bf16_kernel<<<512, 256, 0, stream>>>(Wih[0][1], WCB, 1024 * 128);
  gemm_bf16_kernel<<<512, 128, 0, stream>>>(EMB, WCA, nullptr, XW0, nullptr, 2048, 1024, 128, 0);
  gemm_bf16_kernel<<<512, 128, 0, stream>>>(EMB, WCB, nullptr, XW1, nullptr, 2048, 1024, 128, 0);
  lstm_layer_kernel<<<2, 1024, 0, stream>>>(XW0, XW1, Whh[0][0], Whh[0][1], Bg[0][0], Bg[0][1], X1);

  // ---- LSTM layer 1 ----
  f32_to_bf16_kernel<<<4096, 256, 0, stream>>>(X1, X1B, 2048 * 512);
  f32_to_bf16_kernel<<<2048, 256, 0, stream>>>(Wih[1][0], WCA, 1024 * 512);
  f32_to_bf16_kernel<<<2048, 256, 0, stream>>>(Wih[1][1], WCB, 1024 * 512);
  gemm_bf16_kernel<<<512, 128, 0, stream>>>(X1B, WCA, nullptr, XW0, nullptr, 2048, 1024, 512, 0);
  gemm_bf16_kernel<<<512, 128, 0, stream>>>(X1B, WCB, nullptr, XW1, nullptr, 2048, 1024, 512, 0);
  lstm_layer_kernel<<<2, 1024, 0, stream>>>(XW0, XW1, Whh[1][0], Whh[1][1], Bg[1][0], Bg[1][1], X2);

  // ---- CRF ----
  emission_kernel<<<128, 256, 0, stream>>>(X2, emW, emb_b, EMIS);
  crf_kernel<<<1, 256, 0, stream>>>(EMIS, label_ids, crs, cre, crt, MISC);

  // ---- relation head ----
  relenc_kernel<<<(2048 * 576 + 255) / 256, 256, 0, stream>>>(X2, enttab, label_ids, RENC);
  transpose_bf16_kernel<<<288, 256, 0, stream>>>(suW, SUWT, 576, 128);
  transpose_bf16_kernel<<<288, 256, 0, stream>>>(svW, SVWT, 576, 128);
  transpose_bf16_kernel<<<64, 256, 0, stream>>>(suvW, WAT, 128, 128);
  transpose_bf16_kernel<<<64, 256, 0, stream>>>(suvW + 128 * 128, WBT, 128, 128);
  relprep_kernel<<<32, 256, 0, stream>>>(reltab, RELW);
  gemm_bf16_kernel<<<64, 128, 0, stream>>>(RENC, SUWT, sub, nullptr, UB, 2048, 128, 576, 1);
  gemm_bf16_kernel<<<64, 128, 0, stream>>>(RENC, SVWT, svb, nullptr, VB, 2048, 128, 576, 1);
  gemm_bf16_kernel<<<64, 128, 0, stream>>>(UB, WAT, nullptr, UWA, nullptr, 2048, 128, 128, 0);
  gemm_bf16_kernel<<<64, 128, 0, stream>>>(VB, WBT, nullptr, VWB, nullptr, 2048, 128, 128, 0);
  rel_fused_kernel<<<2048, 256, 0, stream>>>(UWA, VWB, suvb, RELW, spo, MISC + 1);

  combine_kernel<<<1, 1, 0, stream>>>(MISC, (float*)d_out);
}
